// GCN_LSTM_83648783057023
// MI455X (gfx1250) — compile-verified, matrix operand order checked by reading, not run on hardware
//
#include <hip/hip_runtime.h>
#include <hip/hip_bf16.h>
#include <math.h>

typedef __attribute__((ext_vector_type(16))) __bf16 v16bf;
typedef __attribute__((ext_vector_type(8)))  float  v8f;
typedef int v4i __attribute__((vector_size(16)));

#ifndef HAVE_ASYNC_LDS
#  if defined(__has_builtin)
#    if __has_builtin(__builtin_amdgcn_global_load_async_to_lds_b128)
#      define HAVE_ASYNC_LDS 1
#    endif
#  endif
#endif
#ifndef HAVE_ASYNC_LDS
#  define HAVE_ASYNC_LDS 0
#endif

#define BM 128
#define BN 128
#define BK 32

#if HAVE_ASYNC_LDS
__device__ __forceinline__ void async_copy16(const __bf16* g, __bf16* l) {
    __builtin_amdgcn_global_load_async_to_lds_b128(
        (v4i*)(void*)g, (v4i*)(void*)l, 0, 0);
}
__device__ __forceinline__ void wait_async0() {
#  if __has_builtin(__builtin_amdgcn_s_wait_asynccnt)
    __builtin_amdgcn_s_wait_asynccnt(0);
#  else
    asm volatile("s_wait_asynccnt 0x0" ::: "memory");
#  endif
}
#else
__device__ __forceinline__ void wait_async0() {}
#endif

// Stage one K-slab: A row-major [128][32] bf16, B col-major [128 cols][32 k] bf16.
// GUARD=0: all rows in range, async DMA staging (branch-free).
// GUARD=1: zero-pad A rows >= M, synchronous staging.
template<int BTRANS, int GUARD>
__device__ __forceinline__
void stage_tile(const __bf16* __restrict__ A, const __bf16* __restrict__ B,
                __bf16* __restrict__ lA, __bf16* __restrict__ lB,
                int k0, int blockM0, int blockN0, int lda, int ldb, int M,
                const int* __restrict__ gidx, int tid)
{
    // ---- A: 512 x 16B chunks (straight copy)
    #pragma unroll
    for (int it = 0; it < 2; ++it) {
        int chunk = tid + it * 256;
        int r  = chunk >> 2;              // 0..127
        int c8 = (chunk & 3) << 3;        // 0,8,16,24 (bf16 elems)
        const __bf16* gp = A + (long long)(blockM0 + r) * lda + k0 + c8;
        __bf16* lp = &lA[r * BK + c8];
        if (GUARD) {
            uint4 v = {0u, 0u, 0u, 0u};
            if (blockM0 + r < M) v = *reinterpret_cast<const uint4*>(gp);
            *reinterpret_cast<uint4*>(lp) = v;
        } else {
#if HAVE_ASYNC_LDS
            async_copy16(gp, lp);
#else
            *reinterpret_cast<uint4*>(lp) = *reinterpret_cast<const uint4*>(gp);
#endif
        }
    }
    if (BTRANS) {
        // W stored [N x K]: column n of B == row n of W, K-contiguous -> straight copy
        #pragma unroll
        for (int it = 0; it < 2; ++it) {
            int chunk = tid + it * 256;
            int n  = chunk >> 2;          // 0..127
            int c8 = (chunk & 3) << 3;
            const __bf16* gp = B + (long long)(blockN0 + n) * ldb + k0 + c8;
            __bf16* lp = &lB[n * BK + c8];
#if HAVE_ASYNC_LDS
            async_copy16(gp, lp);
#else
            *reinterpret_cast<uint4*>(lp) = *reinterpret_cast<const uint4*>(gp);
#endif
        }
    } else {
        // B row-major [K x N] (opt. gathered rows): transpose into LDS
        #pragma unroll
        for (int it = 0; it < 2; ++it) {
            int chunk = tid + it * 256;
            int kr = chunk >> 4;          // 0..31
            int n8 = (chunk & 15) << 3;   // 0..120
            int srow = k0 + kr;
            if (gidx) srow = gidx[srow];
            uint4 v = *reinterpret_cast<const uint4*>(B + (long long)srow * ldb + blockN0 + n8);
            const __bf16* ev = reinterpret_cast<const __bf16*>(&v);
            #pragma unroll
            for (int j = 0; j < 8; ++j)
                lB[(n8 + j) * BK + kr] = ev[j];
        }
    }
}

// C = act(A @ B), bf16 inputs, f32 accumulate. Optional f32 and/or bf16 outputs.
template<int BTRANS, int GUARD>
__global__ __launch_bounds__(256)
void gemm_bf16_wmma(const __bf16* __restrict__ A, const __bf16* __restrict__ B,
                    float* __restrict__ Cf, __bf16* __restrict__ Cb,
                    int M, int N, int K, int lda, int ldb, int ldc,
                    const int* __restrict__ gatherB,
                    long long sA, long long sB, long long sG, long long sC,
                    int relu)
{
    __shared__ __align__(32) __bf16 lsA[2][BM * BK];
    __shared__ __align__(32) __bf16 lsB[2][BN * BK];

    const int z = blockIdx.z;
    A += (long long)z * sA;
    B += (long long)z * sB;
    const int* gidx = gatherB ? gatherB + (long long)z * sG : nullptr;

    const int tid  = threadIdx.x;
    const int wave = tid >> 5, lane = tid & 31;
    const int half = lane >> 4, l15 = lane & 15;
    const int waveM = wave & 3;           // 4 waves x 32 rows = 128
    const int waveN = wave >> 2;          // 2 waves x 64 cols = 128
    const int blockM0 = blockIdx.x * BM;
    const int blockN0 = blockIdx.y * BN;

    v8f acc[2][4] = {};

    const int niter = K / BK;
    stage_tile<BTRANS, GUARD>(A, B, lsA[0], lsB[0], 0, blockM0, blockN0,
                              lda, ldb, M, gidx, tid);

    for (int i = 0; i < niter; ++i) {
        wait_async0();
        __syncthreads();
        const int cur = i & 1;
        if (i + 1 < niter)
            stage_tile<BTRANS, GUARD>(A, B, lsA[cur ^ 1], lsB[cur ^ 1], (i + 1) * BK,
                                      blockM0, blockN0, lda, ldb, M, gidx, tid);

        v16bf afrag[2], bfrag[4];
        #pragma unroll
        for (int mi = 0; mi < 2; ++mi) {
            int ar = waveM * 32 + mi * 16 + l15;
            afrag[mi] = *reinterpret_cast<const v16bf*>(&lsA[cur][ar * BK + half * 16]);
        }
        #pragma unroll
        for (int ni = 0; ni < 4; ++ni) {
            int bc = waveN * 64 + ni * 16 + l15;
            bfrag[ni] = *reinterpret_cast<const v16bf*>(&lsB[cur][bc * BK + half * 16]);
        }
        #pragma unroll
        for (int mi = 0; mi < 2; ++mi)
            #pragma unroll
            for (int ni = 0; ni < 4; ++ni)
                acc[mi][ni] = __builtin_amdgcn_wmma_f32_16x16x32_bf16(
                    false, afrag[mi], false, bfrag[ni], (short)0, acc[mi][ni], false, false);
    }

    float*  Cfp = Cf ? Cf + (long long)z * sC : nullptr;
    __bf16* Cbp = Cb ? Cb + (long long)z * sC : nullptr;
    #pragma unroll
    for (int mi = 0; mi < 2; ++mi) {
        #pragma unroll
        for (int ni = 0; ni < 4; ++ni) {
            int col = blockN0 + waveN * 64 + ni * 16 + l15;
            #pragma unroll
            for (int r = 0; r < 8; ++r) {
                int row = blockM0 + waveM * 32 + mi * 16 + half * 8 + r;
                if (!GUARD || row < M) {
                    float v = acc[mi][ni][r];
                    if (relu) v = fmaxf(v, 0.f);
                    long long off = (long long)row * ldc + col;
                    if (Cfp) Cfp[off] = v;
                    if (Cbp) Cbp[off] = (__bf16)v;
                }
            }
        }
    }
}

__global__ void cvt_f32_bf16(const float* __restrict__ src, __bf16* __restrict__ dst,
                             long long n) {
    long long i = ((long long)blockIdx.x * blockDim.x + threadIdx.x) * 4;
    if (i + 3 < n) {
        float4 v = *reinterpret_cast<const float4*>(src + i);
        dst[i + 0] = (__bf16)v.x; dst[i + 1] = (__bf16)v.y;
        dst[i + 2] = (__bf16)v.z; dst[i + 3] = (__bf16)v.w;
    }
}

__global__ void inv_init_kernel(int* __restrict__ inv, int n) {
    int i = blockIdx.x * blockDim.x + threadIdx.x;
    if (i < n) inv[i] = -1;
}

__global__ void inv_scatter_kernel(const int* __restrict__ mask, int* __restrict__ inv,
                                   int T, int M, int N) {
    int i = blockIdx.x * blockDim.x + threadIdx.x;
    if (i < T * M) {
        int t = i / M, m = i % M;
        inv[(long long)t * N + mask[i]] = m;
    }
}

__device__ __forceinline__ float sigm(float x) { return 1.0f / (1.0f + __expf(-x)); }

__global__ __launch_bounds__(256)
void lstm_pointwise(const float* __restrict__ zbase, const float* __restrict__ zmask,
                    const float* __restrict__ hh, const int* __restrict__ inv_t,
                    const float* __restrict__ b_ih, const float* __restrict__ b_hh,
                    float* __restrict__ cbuf, __bf16* __restrict__ h_bf,
                    float* __restrict__ h_f32, int H)
{
    int n = blockIdx.x;
    int j = threadIdx.x;                  // H == 256 == blockDim.x
    int m = inv_t[n];
    const float* zr = (m >= 0) ? (zmask + (long long)m * 4 * H)
                               : (zbase + (long long)n * 4 * H);
    const float* hr = hh + (long long)n * 4 * H;
    float zi = zr[0*H + j] + hr[0*H + j] + b_ih[0*H + j] + b_hh[0*H + j];
    float zf = zr[1*H + j] + hr[1*H + j] + b_ih[1*H + j] + b_hh[1*H + j];
    float zg = zr[2*H + j] + hr[2*H + j] + b_ih[2*H + j] + b_hh[2*H + j];
    float zo = zr[3*H + j] + hr[3*H + j] + b_ih[3*H + j] + b_hh[3*H + j];
    long long idx = (long long)n * H + j;
    float c = sigm(zf) * cbuf[idx] + sigm(zi) * tanhf(zg);
    float h = sigm(zo) * tanhf(c);
    cbuf[idx] = c;
    h_bf[idx] = (__bf16)h;
    if (h_f32) h_f32[idx] = h;
}

// Split launch: full 128-row tiles run the branch-free async kernel;
// the M-remainder (if any) runs a single guarded tail block-row.
static inline void launch_gemm(const __bf16* A, const __bf16* B, float* Cf, __bf16* Cb,
                               int M, int N, int K, int lda, int ldb, int ldc,
                               const int* gather, long long sA, long long sB,
                               long long sG, long long sC, int bTrans, int relu,
                               int batch, hipStream_t stream)
{
    int mBlocks = M / BM;
    int mRem    = M - mBlocks * BM;
    if (mBlocks > 0) {
        dim3 grid(mBlocks, N / BN, batch);
        if (bTrans)
            gemm_bf16_wmma<1, 0><<<grid, 256, 0, stream>>>(A, B, Cf, Cb, mBlocks * BM, N, K,
                lda, ldb, ldc, gather, sA, sB, sG, sC, relu);
        else
            gemm_bf16_wmma<0, 0><<<grid, 256, 0, stream>>>(A, B, Cf, Cb, mBlocks * BM, N, K,
                lda, ldb, ldc, gather, sA, sB, sG, sC, relu);
    }
    if (mRem > 0) {
        long long ao = (long long)mBlocks * BM * lda;
        long long co = (long long)mBlocks * BM * ldc;
        dim3 grid(1, N / BN, batch);
        float*  Cf2 = Cf ? Cf + co : nullptr;
        __bf16* Cb2 = Cb ? Cb + co : nullptr;
        if (bTrans)
            gemm_bf16_wmma<1, 1><<<grid, 256, 0, stream>>>(A + ao, B, Cf2, Cb2, mRem, N, K,
                lda, ldb, ldc, gather, sA, sB, sG, sC, relu);
        else
            gemm_bf16_wmma<0, 1><<<grid, 256, 0, stream>>>(A + ao, B, Cf2, Cb2, mRem, N, K,
                lda, ldb, ldc, gather, sA, sB, sG, sC, relu);
    }
}

static inline void launch_cvt(const float* s, __bf16* d, long long n, hipStream_t st) {
    long long blocks = (n / 4 + 255) / 256;
    cvt_f32_bf16<<<(int)blocks, 256, 0, st>>>(s, d, n);
}

extern "C" void kernel_launch(void* const* d_in, const int* in_sizes, int n_in,
                              void* d_out, int out_size, void* d_ws, size_t ws_size,
                              hipStream_t stream)
{
    const float* A_t  = (const float*)d_in[0];   // [16,2048,2048]
    const float* nf   = (const float*)d_in[1];   // [10000,256]
    const int*   mask = (const int*)d_in[2];     // [16,2048]
    const float* w0   = (const float*)d_in[3];   // [256,256]
    const float* w1   = (const float*)d_in[4];   // [256,256]
    const float* Wih  = (const float*)d_in[5];   // [1024,256]
    const float* Whh  = (const float*)d_in[6];   // [1024,256]
    const float* bih  = (const float*)d_in[7];   // [1024]
    const float* bhh  = (const float*)d_in[8];   // [1024]
    float* out = (float*)d_out;

    const int T = 16, N = 10000, M = 2048, F = 256, H = 256, G4 = 1024;

    char* p = (char*)d_ws;
    auto alloc = [&](size_t bytes) -> void* {
        void* r = (void*)p; p += (bytes + 255) & ~(size_t)255; return r;
    };
    __bf16* Abf    = (__bf16*)alloc((size_t)T * M * M * 2);   // 134 MB (mostly L2-resident)
    __bf16* nfb    = (__bf16*)alloc((size_t)N * F * 2);
    __bf16* w0b    = (__bf16*)alloc((size_t)F * F * 2);
    __bf16* w1b    = (__bf16*)alloc((size_t)F * F * 2);
    __bf16* Wihb   = (__bf16*)alloc((size_t)G4 * F * 2);
    __bf16* Whhb   = (__bf16*)alloc((size_t)G4 * F * 2);
    __bf16* base0b = (__bf16*)alloc((size_t)N * F * 2);
    __bf16* base1b = (__bf16*)alloc((size_t)N * F * 2);
    __bf16* Y0b    = (__bf16*)alloc((size_t)T * M * F * 2);
    __bf16* X1b    = (__bf16*)alloc((size_t)T * M * F * 2);
    __bf16* Y1b    = (__bf16*)alloc((size_t)T * M * F * 2);
    __bf16* hb     = (__bf16*)alloc((size_t)N * H * 2);
    float*  zbase  = (float*)alloc((size_t)N * G4 * 4);
    float*  cbuf   = (float*)alloc((size_t)N * H * 4);
    int*    inv    = (int*)alloc((size_t)T * N * 4);
    // LSTM-phase scratch aliased onto Abf (dead after the GCN phase):
    float* Zmask = (float*)Abf;                               // 2048 x 1024 f32 (8 MB)
    float* Hh    = (float*)((char*)Abf + (((size_t)M * G4 * 4 + 255) & ~(size_t)255));

    // ---- pre-convert all GEMM operands to bf16 (one pass each) ----
    launch_cvt(A_t, Abf, (long long)T * M * M, stream);
    launch_cvt(nf,  nfb, (long long)N * F, stream);
    launch_cvt(w0,  w0b, (long long)F * F, stream);
    launch_cvt(w1,  w1b, (long long)F * F, stream);
    launch_cvt(Wih, Wihb, (long long)G4 * F, stream);
    launch_cvt(Whh, Whhb, (long long)G4 * F, stream);

    (void)hipMemsetAsync(hb,   0, (size_t)N * H * 2, stream);
    (void)hipMemsetAsync(cbuf, 0, (size_t)N * H * 4, stream);

    inv_init_kernel<<<(T * N + 255) / 256, 256, 0, stream>>>(inv, T * N);
    inv_scatter_kernel<<<(T * M + 255) / 256, 256, 0, stream>>>(mask, inv, T, M, N);

    // ---- one-time projections ----
    launch_gemm(nfb,    w0b,  nullptr, base0b, N, F,  F, F, F, F, nullptr, 0,0,0,0, 0, 0, 1, stream);
    launch_gemm(base0b, w1b,  nullptr, base1b, N, F,  F, F, F, F, nullptr, 0,0,0,0, 0, 0, 1, stream);
    launch_gemm(base1b, Wihb, zbase,  nullptr, N, G4, F, F, F, G4, nullptr, 0,0,0,0, 1, 0, 1, stream);

    // ---- GCN phase, batched over all 16 timesteps ----
    launch_gemm(Abf, base0b, nullptr, Y0b, M, F, M, M, F, F, mask,
                (long long)M * M, 0, M, (long long)M * F, 0, 1, T, stream);
    launch_gemm(Y0b, w1b, nullptr, X1b, T * M, F, F, F, F, F, nullptr, 0,0,0,0, 0, 0, 1, stream);
    launch_gemm(Abf, X1b, nullptr, Y1b, M, F, M, M, F, F, nullptr,
                (long long)M * M, (long long)M * F, 0, (long long)M * F, 0, 1, T, stream);

    // ---- LSTM phase (sequential over t) ----
    for (int t = 0; t < T; ++t) {
        launch_gemm(Y1b + (long long)t * M * F, Wihb, Zmask, nullptr, M, G4, F, F, F, G4,
                    nullptr, 0,0,0,0, 1, 0, 1, stream);
        launch_gemm(hb, Whhb, Hh, nullptr, N, G4, F, F, F, G4,
                    nullptr, 0,0,0,0, 1, 0, 1, stream);
        lstm_pointwise<<<N, H, 0, stream>>>(zbase, Zmask, Hh, inv + (long long)t * N,
                                            bih, bhh, cbuf, hb,
                                            (t == T - 1) ? out : nullptr, H);
    }
}
